// InteractionBlock_36189394436941
// MI455X (gfx1250) — matrix operand order repaired and checked by reading, not compiled, not run on hardware
//
#include <hip/hip_runtime.h>
#include <hip/hip_bf16.h>

// ---------------------------------------------------------------------------
// DimeNet InteractionBlock for MI455X (gfx1250), bf16 WMMA everywhere.
//   H=128, NB=8, NRAD=6, NSPH*NRAD=42, NBS=1, NAS=2
// ---------------------------------------------------------------------------

typedef __attribute__((ext_vector_type(16))) __bf16 v16bf;
typedef __attribute__((ext_vector_type(8)))  __bf16 v8bf;
typedef __attribute__((ext_vector_type(8)))  float  v8f;
typedef __attribute__((ext_vector_type(4)))  float  v4f;

__device__ __forceinline__ v8f v8f_zero() {
    v8f z = {0.f,0.f,0.f,0.f,0.f,0.f,0.f,0.f};
    return z;
}

// Fast swish: v_exp + v_rcp (trans ops co-execute with WMMA), no IEEE divide.
__device__ __forceinline__ float swish_f(float h) {
    return h * __builtin_amdgcn_rcpf(1.0f + __expf(-h));
}

// Store one bf16 element of a 128x128 B matrix (k = row/K-dim, n = col/N-dim)
// into LDS pre-swizzled to a V_WMMA_*_16X16X32 B-fragment layout in this code's contiguous k order:
// 8 col-tiles x 4 K-chunks, each fragment block = 32 lanes * 32B = 1024B,
// lane = (n%16) + 16*(kk>=16), 16 contiguous bf16 per lane.
__device__ __forceinline__ void b_swizzle_store(char* base, int k, int n, __bf16 v) {
    int ct   = n >> 4;
    int c    = k >> 5;
    int kk   = k & 31;
    int lane = (n & 15) + ((kk >> 4) << 4);
    int pos  = kk & 15;
    *(__bf16*)(base + (((ct << 2) + c) << 10) + lane * 32 + pos * 2) = v;
}

// Load one B fragment (32 contiguous bytes per lane) from swizzled LDS.
__device__ __forceinline__ v16bf load_b_frag(const char* bslot, int ct, int c, int lane) {
    return *(const v16bf*)(bslot + (((ct << 2) + c) << 10) + lane * 32);
}

// Load an A fragment (16x32 bf16, M x K) from a row-major bf16 LDS region
// with 256B row stride.  Layout: lanes 0-15 hold M=lane, K {c*32+0..15};
// lanes 16-31 hold M=lane-16, K {c*32+16..31} (matches swizzled B).
__device__ __forceinline__ v16bf load_a_frag(const char* abase, int rowtile, int c, int lane) {
    int m  = lane & 15;
    int hi = lane >> 4;
    const char* r = abase + (rowtile * 16 + m) * 256;
    int k0 = c * 32 + (hi ? 16 : 0);
    union { v16bf v; v8bf h[2]; } u;
    u.h[0] = *(const v8bf*)(r + k0 * 2);
    u.h[1] = *(const v8bf*)(r + (k0 + 8) * 2);
    return u.v;
}

// ===========================================================================
// Kernel 1: edge transforms.
//   agg[e,:]  = swish(x @ w_ji + b_ji)                      (x_ji, scatter base)
//   xkj[e,:]  = bf16( swish(x @ w_kj + b_kj) * (rbf@w_rbf) )
// 128 edges per block, 256 threads (8 waves), each wave owns one 16-row tile.
// Dynamic LDS: A 32KB | B_ji 32KB | B_kj 32KB | rbf_p 64KB  = 160KB
// ===========================================================================
template<bool FULL>
__device__ __forceinline__ void edge_compute(const char* sA, const char* sBji,
                                             const char* sBkj, const float* sR,
                                             const float* b_ji, const float* b_kj,
                                             float* __restrict__ agg,
                                             __bf16* __restrict__ xkj,
                                             int row0, int E, int rt, int lane) {
    const int hi = (lane >> 4) ? 8 : 0;
    v16bf afr[4];
    #pragma unroll
    for (int c = 0; c < 4; ++c) afr[c] = load_a_frag(sA, rt, c, lane);

    #pragma unroll
    for (int m = 0; m < 2; ++m) {
        const char*  bs   = m ? sBkj : sBji;
        const float* bias = m ? b_kj : b_ji;
        #pragma unroll
        for (int ct = 0; ct < 8; ++ct) {
            v8f acc = v8f_zero();
            #pragma unroll
            for (int c = 0; c < 4; ++c)
                acc = __builtin_amdgcn_wmma_f32_16x16x32_bf16(
                        false, afr[c], false, load_b_frag(bs, ct, c, lane),
                        (short)0, acc, false, false);
            int col = ct * 16 + (lane & 15);
            float bv = bias[col];
            #pragma unroll
            for (int g = 0; g < 8; ++g) {
                int r    = rt * 16 + g + hi;
                int grow = row0 + r;
                if (FULL || grow < E) {
                    float s = swish_f(acc[g] + bv);
                    if (m == 0) agg[(size_t)grow * 128 + col] = s;
                    else        xkj[(size_t)grow * 128 + col] =
                                    (__bf16)(s * sR[r * 128 + col]);
                }
            }
        }
    }
}

__global__ void edge_kernel(const float* __restrict__ x,
                            const float* __restrict__ rbf,
                            const float* __restrict__ w_rbf,
                            const float* __restrict__ w_ji,
                            const float* __restrict__ b_ji,
                            const float* __restrict__ w_kj,
                            const float* __restrict__ b_kj,
                            float* __restrict__ agg,
                            __bf16* __restrict__ xkj,
                            int E) {
    extern __shared__ char smem[];
    char*  sA   = smem;                 // 128 rows x 128 bf16, row stride 256B
    char*  sBji = smem + 32768;
    char*  sBkj = smem + 65536;
    float* sR   = (float*)(smem + 98304); // rbf_p tile, 128x128 f32

    const int tid  = threadIdx.x;
    const int row0 = blockIdx.x * 128;
    const bool full = (row0 + 128) <= E;

    // Fill A (bf16 x tile) and both swizzled B matrices.
    for (int idx = tid; idx < 128 * 128; idx += 256) {
        int a = idx >> 7, b = idx & 127;
        int grow = row0 + a;
        float v = (full || grow < E) ? x[(size_t)grow * 128 + b] : 0.f;
        *(__bf16*)(sA + a * 256 + b * 2) = (__bf16)v;
        b_swizzle_store(sBji, a, b, (__bf16)w_ji[idx]);   // k=a, n=b
        b_swizzle_store(sBkj, a, b, (__bf16)w_kj[idx]);
    }
    // rbf_p tile: [128 rows x 128 cols], K=6 (VALU).
    for (int idx = tid; idx < 128 * 128; idx += 256) {
        int r = idx >> 7, col = idx & 127;
        int grow = row0 + r;
        float acc = 0.f;
        if (full || grow < E) {
            #pragma unroll
            for (int q = 0; q < 6; ++q)
                acc += rbf[(size_t)grow * 6 + q] * w_rbf[q * 128 + col];
        }
        sR[idx] = acc;
    }
    __syncthreads();

    const int wave = tid >> 5;
    const int lane = tid & 31;
    if (full) edge_compute<true >(sA, sBji, sBkj, sR, b_ji, b_kj, agg, xkj, row0, E, wave, lane);
    else      edge_compute<false>(sA, sBji, sBkj, sR, b_ji, b_kj, agg, xkj, row0, E, wave, lane);
}

// ===========================================================================
// Kernel 2: sbf_p[t, 0..7] = sbf[t, 0..41] @ w_sbf  (tiny K, VALU)
// ===========================================================================
__global__ void sbf_kernel(const float* __restrict__ sbf,
                           const float* __restrict__ w_sbf,
                           float* __restrict__ sbf_p, int T) {
    __shared__ float ws[42 * 8];
    int tid = threadIdx.x;
    for (int i = tid; i < 336; i += blockDim.x) ws[i] = w_sbf[i];
    __syncthreads();
    int t = blockIdx.x * blockDim.x + tid;
    if (t >= T) return;
    float acc[8] = {0.f,0.f,0.f,0.f,0.f,0.f,0.f,0.f};
    const float* srow = sbf + (size_t)t * 42;
    #pragma unroll
    for (int r = 0; r < 42; ++r) {
        float s = srow[r];
        #pragma unroll
        for (int j = 0; j < 8; ++j) acc[j] += s * ws[r * 8 + j];
    }
    #pragma unroll
    for (int j = 0; j < 8; ++j) sbf_p[(size_t)t * 8 + j] = acc[j];
}

// ===========================================================================
// Kernel 3: triplet bilinear + scatter.
//   m[w,i] = sum_j sbf_p[w,j] * sum_l xg[w,l] * W[i,j,l];  agg[idx_ji[w]] += m[w]
// Wt_j[l,i] = W[i,j,l] kept bf16-swizzled in LDS (8 x 32KB = 256KB).
// 8 waves = 4 wave-pairs, each pair owns a 16-triplet tile, wave owns 64 cols.
// Dynamic LDS: Wt 256KB | A-stage 4x4KB | sbf-stage 4x512B = 274KB
// ===========================================================================
template<bool FULL>
__device__ __forceinline__ void trip_stage(const __bf16* __restrict__ xkj,
                                           const float*  __restrict__ sbf_p,
                                           const int*    __restrict__ idx_kj,
                                           char* sA, float* sS,
                                           int tile, int T, int NT,
                                           int tl, int nextStride) {
    int r = tl >> 2, seg = tl & 3;
    int w = tile * 16 + r;
    char* dst = sA + r * 256 + seg * 64;
    if (FULL || w < T) {
        int e = idx_kj[w];
        const char* src = (const char*)(xkj + (size_t)e * 128) + seg * 64;
        #pragma unroll
        for (int q = 0; q < 4; ++q)
            *(v4f*)(dst + q * 16) = *(const v4f*)(src + q * 16);
        if (seg == 0) {
            const float* sp = sbf_p + (size_t)w * 8;
            #pragma unroll
            for (int q = 0; q < 8; ++q) sS[r * 8 + q] = sp[q];
        }
    } else {
        v4f z = {0.f,0.f,0.f,0.f};
        #pragma unroll
        for (int q = 0; q < 4; ++q) *(v4f*)(dst + q * 16) = z;
        if (seg == 0) {
            #pragma unroll
            for (int q = 0; q < 8; ++q) sS[r * 8 + q] = 0.f;
        }
    }
    // Prefetch next tile's gathered row (global_prefetch_b8 path).
    int ntile = tile + nextStride;
    if (seg == 0 && ntile < NT) {
        int nw = ntile * 16 + r;
        if (nw < T) {
            int ne = idx_kj[nw];
            __builtin_prefetch((const void*)(xkj + (size_t)ne * 128), 0, 1);
        }
    }
}

__global__ void triplet_kernel(const __bf16* __restrict__ xkj,
                               const float*  __restrict__ sbf_p,
                               const int*    __restrict__ idx_kj,
                               const int*    __restrict__ idx_ji,
                               const float*  __restrict__ W,
                               float* __restrict__ agg,
                               int T, int nIter) {
    extern __shared__ char smem[];
    const int tid  = threadIdx.x;
    const int wave = tid >> 5;
    const int lane = tid & 31;
    const int p    = wave >> 1;      // pair 0..3
    const int half = wave & 1;       // column half
    char*  sA = smem + 262144 + p * 4096;          // 16 rows x 256B
    float* sS = (float*)(smem + 278528 + p * 512); // 16 rows x 8 f32

    // Fill Wt: W flat index = (i*8 + j)*128 + l   ->  matrix j, B[k=l, n=i].
    for (int idx = tid; idx < 8 * 128 * 128; idx += 256) {
        int i = idx >> 10, j = (idx >> 7) & 7, l = idx & 127;
        b_swizzle_store(smem + j * 32768, l, i, (__bf16)W[idx]);
    }
    __syncthreads();

    const int NT = (T + 15) / 16;
    const int hi = (lane >> 4) ? 8 : 0;
    const int nextStride = gridDim.x * 4;

    for (int it = 0; it < nIter; ++it) {
        int tile = (blockIdx.x * 4 + p) + it * nextStride;
        bool active   = tile < NT;
        bool tileFull = (tile * 16 + 16) <= T;
        __syncthreads();            // protect LDS stage vs previous iteration
        if (active) {
            int tl = tid & 63;      // 64 lanes per pair stage 16 rows (256B each)
            if (tileFull) trip_stage<true >(xkj, sbf_p, idx_kj, sA, sS, tile, T, NT, tl, nextStride);
            else          trip_stage<false>(xkj, sbf_p, idx_kj, sA, sS, tile, T, NT, tl, nextStride);
        }
        __syncthreads();
        if (active) {
            v16bf afr[4];
            #pragma unroll
            for (int c = 0; c < 4; ++c) afr[c] = load_a_frag(sA, 0, c, lane);
            v8f acc[4];
            #pragma unroll
            for (int q = 0; q < 4; ++q) acc[q] = v8f_zero();

            #pragma unroll
            for (int j = 0; j < 8; ++j) {
                v8f y[4];
                #pragma unroll
                for (int q = 0; q < 4; ++q) y[q] = v8f_zero();
                #pragma unroll
                for (int ct = 0; ct < 4; ++ct) {
                    int ct4 = half * 4 + ct;
                    #pragma unroll
                    for (int c = 0; c < 4; ++c)
                        y[ct] = __builtin_amdgcn_wmma_f32_16x16x32_bf16(
                                  false, afr[c], false,
                                  load_b_frag(smem + j * 32768, ct4, c, lane),
                                  (short)0, y[ct], false, false);
                }
                float sv[8];
                #pragma unroll
                for (int g = 0; g < 8; ++g) sv[g] = sS[(g + hi) * 8 + j];
                #pragma unroll
                for (int ct = 0; ct < 4; ++ct)
                    #pragma unroll
                    for (int g = 0; g < 8; ++g) acc[ct][g] += sv[g] * y[ct][g];
            }
            // Scatter-add into agg (pre-loaded with x_ji by edge_kernel).
            int nbase = half * 64 + (lane & 15);
            if (tileFull) {
                #pragma unroll
                for (int g = 0; g < 8; ++g) {
                    int w = tile * 16 + g + hi;
                    float* dstp = agg + (size_t)idx_ji[w] * 128 + nbase;
                    #pragma unroll
                    for (int ct = 0; ct < 4; ++ct)
                        atomicAdd(dstp + ct * 16, acc[ct][g]);
                }
            } else {
                #pragma unroll
                for (int g = 0; g < 8; ++g) {
                    int w = tile * 16 + g + hi;
                    if (w < T) {
                        float* dstp = agg + (size_t)idx_ji[w] * 128 + nbase;
                        #pragma unroll
                        for (int ct = 0; ct < 4; ++ct)
                            atomicAdd(dstp + ct * 16, acc[ct][g]);
                    }
                }
            }
        }
    }
}

// ===========================================================================
// Kernel 4: fused residual MLP stack (7 GEMMs, edge rows independent).
//   h = agg; NBS block; h = swish(h@w_lin+b)+x; 2x NAS blocks; out = h
// 64-row tiles; 8 waves = 4 rowtiles x 2 col-halves; residual h in registers;
// bf16 activations round-trip through the LDS A-stage between layers.
// Dynamic LDS: 7 swizzled weights 224KB | biases 3.5KB | A-stage 16KB = 243.5KB
// ===========================================================================
template<bool FULL>
__device__ __forceinline__ void mlp_tile(char* smem, char* sA, const float* sBias,
                                         const float* __restrict__ x,
                                         const float* __restrict__ agg,
                                         float* __restrict__ out,
                                         int row0, int E, int tid,
                                         int rt, int half, int lane) {
    const int mcol = lane & 15;
    const int hi   = (lane >> 4) ? 8 : 0;
    // layer modes: 0 = plain swish->A, 1 = h+=t, 2 = h=t+x, 3 = final h+=t->out
    const int mode7[7] = {0, 1, 2, 0, 1, 0, 3};

    for (int idx = tid; idx < 8192; idx += 256) {
        int r = idx >> 7, col = idx & 127;
        int grow = row0 + r;
        float v = (FULL || grow < E) ? agg[(size_t)grow * 128 + col] : 0.f;
        *(__bf16*)(sA + r * 256 + col * 2) = (__bf16)v;
    }
    float hval[4][8];
    #pragma unroll
    for (int ct = 0; ct < 4; ++ct) {
        int col = half * 64 + ct * 16 + mcol;
        #pragma unroll
        for (int g = 0; g < 8; ++g) {
            int grow = row0 + rt * 16 + g + hi;
            hval[ct][g] = (FULL || grow < E) ? agg[(size_t)grow * 128 + col] : 0.f;
        }
    }
    for (int L = 0; L < 7; ++L) {
        __syncthreads();          // A fully written
        v16bf afr[4];
        #pragma unroll
        for (int c = 0; c < 4; ++c) afr[c] = load_a_frag(sA, rt, c, lane);
        v8f acc[4];
        #pragma unroll
        for (int ct = 0; ct < 4; ++ct) {
            v8f a = v8f_zero();
            int ct4 = half * 4 + ct;
            #pragma unroll
            for (int c = 0; c < 4; ++c)
                a = __builtin_amdgcn_wmma_f32_16x16x32_bf16(
                        false, afr[c], false,
                        load_b_frag(smem + L * 32768, ct4, c, lane),
                        (short)0, a, false, false);
            acc[ct] = a;
        }
        __syncthreads();          // all reads of A done before overwrite
        int md = mode7[L];
        #pragma unroll
        for (int ct = 0; ct < 4; ++ct) {
            int col = half * 64 + ct * 16 + mcol;
            float bv = sBias[L * 128 + col];
            #pragma unroll
            for (int g = 0; g < 8; ++g) {
                int r    = rt * 16 + g + hi;
                int grow = row0 + r;
                float t = swish_f(acc[ct][g] + bv);
                float wv;
                if (md == 0) {
                    wv = t;
                } else if (md == 1) {
                    hval[ct][g] += t; wv = hval[ct][g];
                } else if (md == 2) {
                    float xv = (FULL || grow < E) ? x[(size_t)grow * 128 + col] : 0.f;
                    hval[ct][g] = t + xv; wv = hval[ct][g];
                } else {
                    hval[ct][g] += t; wv = hval[ct][g];
                    if (FULL || grow < E) out[(size_t)grow * 128 + col] = wv;
                }
                *(__bf16*)(sA + r * 256 + col * 2) = (__bf16)wv;
            }
        }
    }
}

__global__ void mlp_kernel(const float* __restrict__ x,
                           const float* __restrict__ agg,
                           const float* __restrict__ rb_w1, const float* __restrict__ rb_b1,
                           const float* __restrict__ rb_w2, const float* __restrict__ rb_b2,
                           const float* __restrict__ w_lin, const float* __restrict__ b_lin,
                           const float* __restrict__ ra_w1, const float* __restrict__ ra_b1,
                           const float* __restrict__ ra_w2, const float* __restrict__ ra_b2,
                           float* __restrict__ out,
                           int E, int nIter) {
    extern __shared__ char smem[];
    float* sBias = (float*)(smem + 229376);
    char*  sA    = smem + 232960;   // 64 rows x 256B bf16

    const int tid = threadIdx.x;
    const float* wsrc[7] = {rb_w1, rb_w2, w_lin, ra_w1, ra_w2, ra_w1 + 16384, ra_w2 + 16384};
    const float* bsrc[7] = {rb_b1, rb_b2, b_lin, ra_b1, ra_b2, ra_b1 + 128,   ra_b2 + 128};

    for (int s = 0; s < 7; ++s) {
        char* slot = smem + s * 32768;
        const float* wp = wsrc[s];
        for (int idx = tid; idx < 16384; idx += 256)
            b_swizzle_store(slot, idx >> 7, idx & 127, (__bf16)wp[idx]);
    }
    for (int i = tid; i < 896; i += 256) sBias[i] = bsrc[i >> 7][i & 127];
    __syncthreads();

    const int wave = tid >> 5;
    const int lane = tid & 31;
    const int rt   = wave & 3;       // row tile 0..3
    const int half = wave >> 2;      // column half
    const int NT   = (E + 63) / 64;

    for (int it = 0; it < nIter; ++it) {
        int tile = blockIdx.x + it * gridDim.x;   // block-uniform
        if (tile >= NT) break;
        int row0 = tile * 64;
        __syncthreads();   // protect previous iteration's A reads vs restaging
        if ((row0 + 64) <= E)
            mlp_tile<true >(smem, sA, sBias, x, agg, out, row0, E, tid, rt, half, lane);
        else
            mlp_tile<false>(smem, sA, sBias, x, agg, out, row0, E, tid, rt, half, lane);
    }
}

// ===========================================================================
// Host launch
// ===========================================================================
extern "C" void kernel_launch(void* const* d_in, const int* in_sizes, int n_in,
                              void* d_out, int out_size, void* d_ws, size_t ws_size,
                              hipStream_t stream) {
    const float* x      = (const float*)d_in[0];
    const float* rbf    = (const float*)d_in[1];
    const float* sbf    = (const float*)d_in[2];
    const int*   idx_kj = (const int*)  d_in[3];
    const int*   idx_ji = (const int*)  d_in[4];
    const float* w_rbf  = (const float*)d_in[5];
    const float* w_sbf  = (const float*)d_in[6];
    const float* w_kj   = (const float*)d_in[7];
    const float* b_kj   = (const float*)d_in[8];
    const float* w_ji   = (const float*)d_in[9];
    const float* b_ji   = (const float*)d_in[10];
    const float* W      = (const float*)d_in[11];
    const float* rb_w1  = (const float*)d_in[12];
    const float* rb_b1  = (const float*)d_in[13];
    const float* rb_w2  = (const float*)d_in[14];
    const float* rb_b2  = (const float*)d_in[15];
    const float* w_lin  = (const float*)d_in[16];
    const float* b_lin  = (const float*)d_in[17];
    const float* ra_w1  = (const float*)d_in[18];
    const float* ra_b1  = (const float*)d_in[19];
    const float* ra_w2  = (const float*)d_in[20];
    const float* ra_b2  = (const float*)d_in[21];

    const int E = in_sizes[0] / 128;
    const int T = in_sizes[3];

    // Workspace carve: agg (E*128 f32) | xkj (E*128 bf16) | sbf_p (T*8 f32)
    char*   ws    = (char*)d_ws;
    size_t  aggB  = (size_t)E * 128 * sizeof(float);
    size_t  xkjB  = (size_t)E * 128 * sizeof(__bf16);
    float*  agg   = (float*)ws;
    __bf16* xkj   = (__bf16*)(ws + aggB);
    float*  sbf_p = (float*)(ws + aggB + xkjB);

    const size_t edgeLds = 163840;   // 160 KB
    const size_t tripLds = 280576;   // 274 KB
    const size_t mlpLds  = 249344;   // 243.5 KB
    (void)hipFuncSetAttribute((const void*)edge_kernel,
        hipFuncAttributeMaxDynamicSharedMemorySize, (int)edgeLds);
    (void)hipFuncSetAttribute((const void*)triplet_kernel,
        hipFuncAttributeMaxDynamicSharedMemorySize, (int)tripLds);
    (void)hipFuncSetAttribute((const void*)mlp_kernel,
        hipFuncAttributeMaxDynamicSharedMemorySize, (int)mlpLds);

    // 1) edge projections: x_ji -> agg, x_kj -> bf16 buffer
    edge_kernel<<<(E + 127) / 128, 256, edgeLds, stream>>>(
        x, rbf, w_rbf, w_ji, b_ji, w_kj, b_kj, agg, xkj, E);

    // 2) sbf projection
    sbf_kernel<<<(T + 255) / 256, 256, 0, stream>>>(sbf, w_sbf, sbf_p, T);

    // 3) triplet bilinear + atomic scatter into agg
    const int tgrid = 256;
    const int NTt   = (T + 15) / 16;
    const int tIter = (NTt + tgrid * 4 - 1) / (tgrid * 4);
    triplet_kernel<<<tgrid, 256, tripLds, stream>>>(
        xkj, sbf_p, idx_kj, idx_ji, W, agg, T, tIter);

    // 4) fused residual MLP stack -> out
    const int mgrid = 256;
    const int NTe   = (E + 63) / 64;
    const int mIter = (NTe + mgrid - 1) / mgrid;
    mlp_kernel<<<mgrid, 256, mlpLds, stream>>>(
        x, agg, rb_w1, rb_b1, rb_w2, rb_b2, w_lin, b_lin,
        ra_w1, ra_b1, ra_w2, ra_b2, (float*)d_out, E, mIter);
}
